// cnf_wrapper_85736137163134
// MI455X (gfx1250) — matrix operand order repaired and checked by reading, not compiled, not run on hardware
//
#include <hip/hip_runtime.h>

typedef __attribute__((ext_vector_type(16))) _Float16 v16h;
typedef __attribute__((ext_vector_type(8)))  float    v8f;

struct Frag { unsigned r[8]; };

__device__ __forceinline__ v16h frag_v(const Frag& f) {
  union { Frag f; v16h v; } u; u.f = f; return u.v;
}

__device__ __forceinline__ unsigned pkh2(float a, float b) {
  union { _Float16 h[2]; unsigned u; } x;
  x.h[0] = (_Float16)a; x.h[1] = (_Float16)b; return x.u;
}

__device__ __forceinline__ unsigned swap16(unsigned v) {
  return (unsigned)__shfl_xor((int)v, 16, 32);
}

__device__ __forceinline__ v8f wmma_f16(const Frag& a, const Frag& b, v8f c) {
  return __builtin_amdgcn_wmma_f32_16x16x32_f16(
      false, frag_v(a), false, frag_v(b), (short)0, c, false, false);
}

// Build next-layer B fragment (32x16 K x N) from two 16x16 f32 D tiles in the
// WMMA C layout. tA = K rows 0..15, tB = K rows 16..31 of the fragment.
__device__ __forceinline__ void pack_B(const float (&tA)[8], const float (&tB)[8],
                                       bool hi, Frag& B) {
#pragma unroll
  for (int j = 0; j < 4; ++j) {
    unsigned p0 = pkh2(tA[2*j], tA[2*j+1]);
    unsigned p1 = pkh2(tB[2*j], tB[2*j+1]);
    unsigned s0 = swap16(p0);
    unsigned s1 = swap16(p1);
    B.r[j]     = p0;
    B.r[4 + j] = p1;
  }
}

// A fragment (16x32 f16) from a row-major 64x64 f32 weight matrix.
__device__ __forceinline__ void load_Afrag(const float* __restrict__ W, int row,
                                           int kbase, int hi8, Frag& f) {
  const float4* p0 = (const float4*)(W + row * 64 + kbase + hi8);
  const float4* p1 = (const float4*)(W + row * 64 + kbase + 16 + hi8);
  float4 a = p0[0], b = p0[1], c = p1[0], d = p1[1];
  f.r[0] = pkh2(a.x, a.y); f.r[1] = pkh2(a.z, a.w);
  f.r[2] = pkh2(b.x, b.y); f.r[3] = pkh2(b.z, b.w);
  f.r[4] = pkh2(c.x, c.y); f.r[5] = pkh2(c.z, c.w);
  f.r[6] = pkh2(d.x, d.y); f.r[7] = pkh2(d.z, d.w);
}

__device__ __forceinline__ Frag lds_frag(const uint4* __restrict__ A0,
                                         const uint4* __restrict__ A1, int idx) {
  union { uint4 q[2]; Frag f; } u;
  u.q[0] = A0[idx];
  u.q[1] = A1[idx];
  return u.f;
}

#define SELU_SC 1.0507009873554805f
#define SELU_SA 1.7580993408473766f   /* scale * alpha */

// h = selu(a), d = selu'(a): t = SA*exp(a); d = a>0 ? SC : t; h = a>0 ? SC*a : t-SA
__device__ __forceinline__ void selu_hd(float a, float& h, float& d) {
  float t = SELU_SA * __expf(a);
  bool pos = a > 0.0f;
  d = pos ? SELU_SC : t;
  h = pos ? SELU_SC * a : t - SELU_SA;
}

__global__ __launch_bounds__(256) void cnf_kernel(
    const float* __restrict__ T,  const float* __restrict__ X,
    const float* __restrict__ W1, const float* __restrict__ b1,
    const float* __restrict__ W2, const float* __restrict__ b2,
    const float* __restrict__ W3, const float* __restrict__ b3,
    const float* __restrict__ W4, const float* __restrict__ b4,
    float* __restrict__ OUT, int ntiles)
{
  // Prepacked weight A-fragments: 16 frags (layer,itile,khalf) x 32 lanes,
  // split into two 16B planes so reloads are two clean ds_load_b128.
  __shared__ uint4  AFR0[16 * 32];
  __shared__ uint4  AFR1[16 * 32];
  __shared__ float2 w1c[64];   // (W1[i,0], W1[i,1])
  __shared__ float2 w4p[64];   // (W4[0,i], W4[1,i])
  __shared__ float  lb2[64];
  __shared__ float  lb3[64];

  const int  tid  = threadIdx.x;
  const int  lane = tid & 31;
  const int  wv   = tid >> 5;
  const int  lm   = lane & 15;
  const int  hi8  = (lane >> 4) * 8;
  const bool hi   = lane >= 16;

  if (tid < 64) {
    w1c[tid] = make_float2(W1[tid * 3 + 0], W1[tid * 3 + 1]);
    w4p[tid] = make_float2(W4[tid], W4[64 + tid]);
    lb2[tid] = b2[tid];
    lb3[tid] = b3[tid];
  }

  // Each of the 8 waves packs 2 of the 16 weight fragments.
#pragma unroll
  for (int s = 0; s < 2; ++s) {
    int f     = wv * 2 + s;            // f = layer*8 + it*2 + kh
    int layer = f >> 3;
    int it    = (f & 7) >> 1;
    int kh    = f & 1;
    Frag fr;
    load_Afrag(layer ? W3 : W2, it * 16 + lm, kh * 32, hi8, fr);
    union { Frag fr; uint4 q[2]; } u; u.fr = fr;
    AFR0[f * 32 + lane] = u.q[0];
    AFR1[f * 32 + lane] = u.q[1];
  }
  __syncthreads();

  const float tval = T[0];
  const float b40 = b4[0], b41 = b4[1];

  // Layer-1 A fragments: K=0..3 = (W1[:,0], W1[:,1], W1[:,2], b1); rest zero.
  unsigned A1r0[4], A1r1[4];
#pragma unroll
  for (int it = 0; it < 4; ++it) {
    int row = it * 16 + lm;
    unsigned r0 = pkh2(W1[row * 3 + 0], W1[row * 3 + 1]);
    unsigned r1 = pkh2(W1[row * 3 + 2], b1[row]);
    A1r0[it] = hi ? 0u : r0;
    A1r1[it] = hi ? 0u : r1;
  }

  const int gwave = (int)((blockIdx.x * blockDim.x + tid) >> 5);
  const int nwave = (int)((gridDim.x * blockDim.x) >> 5);

  for (int tile = gwave; tile < ntiles; tile += nwave) {
    const int p = tile * 16 + lm;

    // B1 fragment: K=0..3 = (x0, x1, t, 1), rest zero.
    Frag B1;
    {
      float x0 = X[p * 3 + 0], x1 = X[p * 3 + 1];
      B1.r[0] = hi ? 0u : pkh2(x0, x1);
      B1.r[1] = hi ? 0u : pkh2(tval, 1.0f);
#pragma unroll
      for (int j = 2; j < 8; ++j) B1.r[j] = 0u;
    }

    // ---- Layer 1: H pass (bias folded into WMMA via constant-1 input) ----
    Frag BH[2];
    float d1[4][8];
    {
      float st[2][8];
#pragma unroll
      for (int itp = 0; itp < 2; ++itp) {
#pragma unroll
        for (int k = 0; k < 2; ++k) {
          int it = itp * 2 + k;
          Frag A1f;
          A1f.r[0] = A1r0[it]; A1f.r[1] = A1r1[it];
#pragma unroll
          for (int j = 2; j < 8; ++j) A1f.r[j] = 0u;
          v8f acc = {};
          acc = wmma_f16(A1f, B1, acc);
#pragma unroll
          for (int r = 0; r < 8; ++r) {
            float h, d;
            selu_hd(acc[r], h, d);
            st[k][r]  = h;
            d1[it][r] = d;
          }
        }
        pack_B(st[0], st[1], hi, BH[itp]);
      }
    }
    // ---- Layer 1: U/V tangent seeds (elementwise, no GEMM needed) ----
    Frag BU[2], BV[2];
    {
      float su[2][8], sv[2][8];
#pragma unroll
      for (int itp = 0; itp < 2; ++itp) {
#pragma unroll
        for (int k = 0; k < 2; ++k) {
          int it = itp * 2 + k;
#pragma unroll
          for (int r = 0; r < 8; ++r) {
            float2 wc = w1c[it * 16 + hi8 + r];
            su[k][r] = d1[it][r] * wc.x;
            sv[k][r] = d1[it][r] * wc.y;
          }
        }
        pack_B(su[0], su[1], hi, BU[itp]);
        pack_B(sv[0], sv[1], hi, BV[itp]);
      }
    }

    // ---- Layer 2: H pass ----
    Frag BH2[2];
    float d2[4][8];
    {
      float st[2][8];
#pragma unroll
      for (int itp = 0; itp < 2; ++itp) {
#pragma unroll
        for (int k = 0; k < 2; ++k) {
          int it = itp * 2 + k;
          Frag A0  = lds_frag(AFR0, AFR1, (it * 2 + 0) * 32 + lane);
          Frag A1k = lds_frag(AFR0, AFR1, (it * 2 + 1) * 32 + lane);
          v8f ah = {};
          ah = wmma_f16(A0, BH[0], ah); ah = wmma_f16(A1k, BH[1], ah);
#pragma unroll
          for (int r = 0; r < 8; ++r) {
            float h, d;
            selu_hd(ah[r] + lb2[it * 16 + hi8 + r], h, d);
            st[k][r]  = h;
            d2[it][r] = d;
          }
        }
        pack_B(st[0], st[1], hi, BH2[itp]);
      }
    }
    // ---- Layer 2: U pass ----
    Frag BU2[2];
    {
      float st[2][8];
#pragma unroll
      for (int itp = 0; itp < 2; ++itp) {
#pragma unroll
        for (int k = 0; k < 2; ++k) {
          int it = itp * 2 + k;
          Frag A0  = lds_frag(AFR0, AFR1, (it * 2 + 0) * 32 + lane);
          Frag A1k = lds_frag(AFR0, AFR1, (it * 2 + 1) * 32 + lane);
          v8f au = {};
          au = wmma_f16(A0, BU[0], au); au = wmma_f16(A1k, BU[1], au);
#pragma unroll
          for (int r = 0; r < 8; ++r) st[k][r] = d2[it][r] * au[r];
        }
        pack_B(st[0], st[1], hi, BU2[itp]);
      }
    }
    // ---- Layer 2: V pass ----
    Frag BV2[2];
    {
      float st[2][8];
#pragma unroll
      for (int itp = 0; itp < 2; ++itp) {
#pragma unroll
        for (int k = 0; k < 2; ++k) {
          int it = itp * 2 + k;
          Frag A0  = lds_frag(AFR0, AFR1, (it * 2 + 0) * 32 + lane);
          Frag A1k = lds_frag(AFR0, AFR1, (it * 2 + 1) * 32 + lane);
          v8f av = {};
          av = wmma_f16(A0, BV[0], av); av = wmma_f16(A1k, BV[1], av);
#pragma unroll
          for (int r = 0; r < 8; ++r) st[k][r] = d2[it][r] * av[r];
        }
        pack_B(st[0], st[1], hi, BV2[itp]);
      }
    }

    // ---- Layer 3 + output layer, stream by stream ----
    float d3[4][8];
    float acc0 = 0.0f, acc1 = 0.0f, accd = 0.0f;
#pragma unroll
    for (int it = 0; it < 4; ++it) {   // H pass: dx accumulation + stash d3
      Frag A0  = lds_frag(AFR0, AFR1, (8 + it * 2 + 0) * 32 + lane);
      Frag A1k = lds_frag(AFR0, AFR1, (8 + it * 2 + 1) * 32 + lane);
      v8f ah = {};
      ah = wmma_f16(A0, BH2[0], ah); ah = wmma_f16(A1k, BH2[1], ah);
#pragma unroll
      for (int r = 0; r < 8; ++r) {
        int   irow = it * 16 + hi8 + r;
        float h3, d;
        selu_hd(ah[r] + lb3[irow], h3, d);
        d3[it][r] = d;
        float2 w4 = w4p[irow];
        acc0 += w4.x * h3;
        acc1 += w4.y * h3;
      }
    }
#pragma unroll
    for (int it = 0; it < 4; ++it) {   // U pass: divergence, d out0/dy0
      Frag A0  = lds_frag(AFR0, AFR1, (8 + it * 2 + 0) * 32 + lane);
      Frag A1k = lds_frag(AFR0, AFR1, (8 + it * 2 + 1) * 32 + lane);
      v8f au = {};
      au = wmma_f16(A0, BU2[0], au); au = wmma_f16(A1k, BU2[1], au);
#pragma unroll
      for (int r = 0; r < 8; ++r) {
        int irow = it * 16 + hi8 + r;
        accd += w4p[irow].x * (d3[it][r] * au[r]);
      }
    }
#pragma unroll
    for (int it = 0; it < 4; ++it) {   // V pass: divergence, d out1/dy1
      Frag A0  = lds_frag(AFR0, AFR1, (8 + it * 2 + 0) * 32 + lane);
      Frag A1k = lds_frag(AFR0, AFR1, (8 + it * 2 + 1) * 32 + lane);
      v8f av = {};
      av = wmma_f16(A0, BV2[0], av); av = wmma_f16(A1k, BV2[1], av);
#pragma unroll
      for (int r = 0; r < 8; ++r) {
        int irow = it * 16 + hi8 + r;
        accd += w4p[irow].y * (d3[it][r] * av[r]);
      }
    }

    acc0 += __shfl_xor(acc0, 16, 32);
    acc1 += __shfl_xor(acc1, 16, 32);
    accd += __shfl_xor(accd, 16, 32);
    if (!hi) {
      OUT[p * 3 + 0] = acc0 + b40;
      OUT[p * 3 + 1] = acc1 + b41;
      OUT[p * 3 + 2] = accd;
    }
  }
}

extern "C" void kernel_launch(void* const* d_in, const int* in_sizes, int n_in,
                              void* d_out, int out_size, void* d_ws, size_t ws_size,
                              hipStream_t stream) {
  const float* T  = (const float*)d_in[0];
  const float* X  = (const float*)d_in[1];
  const float* W1 = (const float*)d_in[2];
  const float* b1 = (const float*)d_in[3];
  const float* W2 = (const float*)d_in[4];
  const float* b2 = (const float*)d_in[5];
  const float* W3 = (const float*)d_in[6];
  const float* b3 = (const float*)d_in[7];
  const float* W4 = (const float*)d_in[8];
  const float* b4 = (const float*)d_in[9];
  float* OUT = (float*)d_out;

  int npts   = in_sizes[1] / 3;
  int ntiles = npts / 16;

  dim3 grid(256), block(256);   // 2048 waves -> 8 tiles/wave at B=262144
  hipLaunchKernelGGL(cnf_kernel, grid, block, 0, stream,
                     T, X, W1, b1, W2, b2, W3, b3, W4, b4, OUT, ntiles);
}
